// LearnableUpsampler_81269371175491
// MI455X (gfx1250) — matrix operand order repaired and checked by reading, not compiled, not run on hardware
//
#include <hip/hip_runtime.h>

#define Bn 16
#define Ts 256
#define Dd 192
#define Tm 1000
#define TmP 1008
#define CO 8
#define KQ 1024   /* Ts*4 : fused K dimension of the big GEMM */
#define KC 64     /* K-chunk staged into LDS by the TDM */
#define EPSL 1e-5f

typedef __attribute__((ext_vector_type(16))) __bf16 v16bf;
typedef __attribute__((ext_vector_type(8)))  float  v8f;
typedef __attribute__((ext_vector_type(4)))  unsigned v4u;
typedef __attribute__((ext_vector_type(8)))  unsigned v8u;

union FragU { v16bf v; unsigned u[8]; };

__device__ __forceinline__ unsigned short f2bf(float f) {
  unsigned u = __float_as_uint(f);
  u += 0x7fffu + ((u >> 16) & 1u);          // round-to-nearest-even
  return (unsigned short)(u >> 16);
}

__device__ __forceinline__ float siluf(float x) { return x / (1.f + __expf(-x)); }

// A-operand: 16x32 bf16, row-major [M, K] with K contiguous (ISA 7.12.2 layout)
__device__ __forceinline__ v16bf load_fragA(const unsigned short* base, int lda,
                                            int lane, int kbase) {
  FragU f;
  int row = lane & 15, kh = lane >> 4;
  const unsigned short* p = base + (size_t)row * lda + kbase + 8 * kh;
  f.u[0] = *(const unsigned*)(p + 0);
  f.u[1] = *(const unsigned*)(p + 2);
  f.u[2] = *(const unsigned*)(p + 4);
  f.u[3] = *(const unsigned*)(p + 6);
  f.u[4] = *(const unsigned*)(p + 16);
  f.u[5] = *(const unsigned*)(p + 18);
  f.u[6] = *(const unsigned*)(p + 20);
  f.u[7] = *(const unsigned*)(p + 22);
  return f.v;
}

// B-operand: 32x16 bf16, stored as [N rows, K contiguous] (i.e. B^T row-major)
__device__ __forceinline__ v16bf load_fragB(const unsigned short* base, int ldb,
                                            int lane, int kbase) {
  FragU f;
  int col = lane & 15, kh = lane >> 4;
  const unsigned short* p = base + (size_t)col * ldb + kbase + 8 * kh;
#pragma unroll
  for (int v = 0; v < 8; ++v) f.u[v] = *(const unsigned*)(p + 2 * v + (v >= 4 ? 8 : 0));
  return f.v;
}

__device__ __forceinline__ v8f wmma_bf16(v16bf a, v16bf b, v8f c) {
  return __builtin_amdgcn_wmma_f32_16x16x32_bf16(false, a, false, b,
                                                 (short)0, c, false, false);
}

// ---------------------------------------------------------------------------
// K1: cumsum (LDS scan), mel_len/mel_mask, both conv blocks, V -> bf16
// ---------------------------------------------------------------------------
__global__ void prep_kernel(
    const float* __restrict__ dur, const float* __restrict__ V,
    const float* cwW, const float* cwB, const float* cwM, const float* cwVv,
    const float* cwG, const float* cwBB, const float* cwLG, const float* cwLB,
    const float* ccW, const float* ccB, const float* ccM, const float* ccVv,
    const float* ccG, const float* ccBB, const float* ccLG, const float* ccLB,
    float* e_k, float* s_k, int* mel_len_i, float* VcW, float* VcC,
    unsigned short* Vbf, float* outMask, float* outMlen)
{
  int b = blockIdx.x, k = threadIdx.x;
  __shared__ float scan[Ts];
  __shared__ float wlA[CO * Dd * 3];
  __shared__ float wlB[CO * Dd * 3];
  __shared__ int mlSh;

  float d = dur[b * Ts + k];
  scan[k] = d;
  __syncthreads();
  for (int off = 1; off < Ts; off <<= 1) {
    float v = scan[k] + ((k >= off) ? scan[k - off] : 0.f);
    __syncthreads();
    scan[k] = v;
    __syncthreads();
  }
  float e = scan[k];
  e_k[b * Ts + k] = e;
  s_k[b * Ts + k] = e - d;
  if (k == 0) {
    int ml = (int)rintf(scan[Ts - 1]);
    ml = ml < 0 ? 0 : (ml > Tm ? Tm : ml);
    mel_len_i[b] = ml;
    outMlen[b] = (float)ml;
    mlSh = ml;
  }
  for (int i = k; i < CO * Dd * 3; i += Ts) { wlA[i] = cwW[i]; wlB[i] = ccW[i]; }
  __syncthreads();
  int ml = mlSh;
  for (int t = k; t < Tm; t += Ts) outMask[b * Tm + t] = (t >= ml) ? 1.f : 0.f;

  float accA[CO], accB[CO];
#pragma unroll
  for (int co = 0; co < CO; ++co) { accA[co] = cwB[co]; accB[co] = ccB[co]; }
  for (int tap = 0; tap < 3; ++tap) {
    int kk = k + tap - 1;
    if (kk < 0 || kk >= Ts) continue;
    const float* vr = V + ((size_t)b * Ts + kk) * Dd;
    for (int c = 0; c < Dd; ++c) {
      float v = vr[c];
      int wi = c * 3 + tap;
#pragma unroll
      for (int co = 0; co < CO; ++co) {
        accA[co] += v * wlA[co * (Dd * 3) + wi];
        accB[co] += v * wlB[co * (Dd * 3) + wi];
      }
    }
  }
  {
    float y[CO]; float mu = 0.f;
#pragma unroll
    for (int co = 0; co < CO; ++co) {
      float t0 = (accA[co] - cwM[co]) * rsqrtf(cwVv[co] + EPSL);
      t0 = siluf(t0 * cwG[co] + cwBB[co]);
      y[co] = t0; mu += t0;
    }
    mu *= (1.f / CO);
    float var = 0.f;
#pragma unroll
    for (int co = 0; co < CO; ++co) { float dd = y[co] - mu; var += dd * dd; }
    float rs = rsqrtf(var * (1.f / CO) + EPSL);
#pragma unroll
    for (int co = 0; co < CO; ++co)
      VcW[((size_t)b * Ts + k) * CO + co] = (y[co] - mu) * rs * cwLG[co] + cwLB[co];
  }
  {
    float y[CO]; float mu = 0.f;
#pragma unroll
    for (int co = 0; co < CO; ++co) {
      float t0 = (accB[co] - ccM[co]) * rsqrtf(ccVv[co] + EPSL);
      t0 = siluf(t0 * ccG[co] + ccBB[co]);
      y[co] = t0; mu += t0;
    }
    mu *= (1.f / CO);
    float var = 0.f;
#pragma unroll
    for (int co = 0; co < CO; ++co) { float dd = y[co] - mu; var += dd * dd; }
    float rs = rsqrtf(var * (1.f / CO) + EPSL);
#pragma unroll
    for (int co = 0; co < CO; ++co)
      VcC[((size_t)b * Ts + k) * CO + co] = (y[co] - mu) * rs * ccLG[co] + ccLB[co];
  }
  for (int i = k; i < Ts * Dd; i += Ts)
    Vbf[(size_t)b * Ts * Dd + i] = f2bf(V[(size_t)b * Ts * Dd + i]);
}

// ---------------------------------------------------------------------------
// K1b: weight matrices -> bf16 (both already [N, K-contiguous] for B-fragments)
// ---------------------------------------------------------------------------
__global__ void cvt_w_kernel(const float* lw_w, const float* po_w,
                             unsigned short* lwbf, unsigned short* pobf) {
  int tid = blockIdx.x * blockDim.x + threadIdx.x;
  int stride = blockDim.x * gridDim.x;
  for (int i = tid; i < Dd * 4 * Dd; i += stride) lwbf[i] = f2bf(lw_w[i]);
  for (int i = tid; i < Dd * Dd; i += stride)     pobf[i] = f2bf(po_w[i]);
}

// ---------------------------------------------------------------------------
// K2: per-(b,t) block: tiny MLPs, block softmax over Ts, Bc via ds_add_f32,
//     writes W (f32 output) and Wflat (bf16 GEMM A-operand)
// ---------------------------------------------------------------------------
__global__ void mlp_kernel(
    const float* __restrict__ e_k, const float* __restrict__ s_k,
    const int* __restrict__ mel_len_i, const int* __restrict__ src_len,
    const float* __restrict__ VcW, const float* __restrict__ VcC,
    const float* swW1, const float* swB1, const float* swW2, const float* swB2,
    const float* swW3, const float* swB3,
    const float* scW1, const float* scB1, const float* scW2, const float* scB2,
    const float* scW3, const float* scB3,
    float* outW, unsigned short* Wbf, float* Bc)
{
  int t = blockIdx.x, b = blockIdx.y, k = threadIdx.x;
  __shared__ float sp[128];
  __shared__ float red[Ts];
  __shared__ float BcS[8];
  __shared__ int mlSh;

  if (k < 40)       sp[k] = swW1[k];
  else if (k < 44)  sp[k] = swB1[k - 40];
  else if (k < 60)  sp[k] = swW2[k - 44];
  else if (k < 64)  sp[k] = swB2[k - 60];
  else if (k < 80)  sp[k] = swW3[k - 64];
  else if (k < 84)  sp[k] = swB3[k - 80];
  else if (k < 104) sp[k] = scW1[k - 84];
  else if (k < 106) sp[k] = scB1[k - 104];
  else if (k < 110) sp[k] = scW2[k - 106];
  else if (k < 112) sp[k] = scB2[k - 110];
  else if (k < 116) sp[k] = scW3[k - 112];
  else if (k < 118) sp[k] = scB3[k - 116];
  if (k == 0) mlSh = mel_len_i[b];
  if (k < 8) BcS[k] = 0.f;
  __syncthreads();
  const float *W1 = sp,       *B1 = sp + 40, *W2 = sp + 44, *B2 = sp + 60;
  const float *W3 = sp + 64,  *B3 = sp + 80;
  const float *C1 = sp + 84,  *Cb1 = sp + 104, *C2 = sp + 106, *Cb2 = sp + 110;
  const float *C3 = sp + 112, *Cb3 = sp + 116;

  bool melv = t < mlSh;
  bool srcv = k < src_len[b];
  bool am = !(melv && srcv);
  float S = am ? 0.f : ((float)(t + 1) - s_k[b * Ts + k]);
  float E = am ? 0.f : (e_k[b * Ts + k] - (float)(t + 1));

  float x[10]; x[0] = S; x[1] = E;
  {
    const float* vw = VcW + ((size_t)b * Ts + k) * CO;
#pragma unroll
    for (int j = 0; j < 8; ++j) x[2 + j] = vw[j];
  }
  float l[4];
  {
    float h[4], g[4];
#pragma unroll
    for (int j = 0; j < 4; ++j) { float a = B1[j]; for (int i = 0; i < 10; ++i) a += W1[j * 10 + i] * x[i]; h[j] = siluf(a); }
#pragma unroll
    for (int j = 0; j < 4; ++j) { float a = B2[j]; for (int i = 0; i < 4; ++i) a += W2[j * 4 + i] * h[i]; g[j] = siluf(a); }
#pragma unroll
    for (int j = 0; j < 4; ++j) { float a = B3[j]; for (int i = 0; i < 4; ++i) a += W3[j * 4 + i] * g[i]; l[j] = a; }
  }
  float c[2];
  {
    float x2[10]; x2[0] = S; x2[1] = E;
    const float* vc = VcC + ((size_t)b * Ts + k) * CO;
#pragma unroll
    for (int j = 0; j < 8; ++j) x2[2 + j] = vc[j];
    float h[2], g[2];
#pragma unroll
    for (int j = 0; j < 2; ++j) { float a = Cb1[j]; for (int i = 0; i < 10; ++i) a += C1[j * 10 + i] * x2[i]; h[j] = siluf(a); }
#pragma unroll
    for (int j = 0; j < 2; ++j) { float a = Cb2[j]; for (int i = 0; i < 2; ++i) a += C2[j * 2 + i] * h[i]; g[j] = siluf(a); }
#pragma unroll
    for (int j = 0; j < 2; ++j) { float a = Cb3[j]; for (int i = 0; i < 2; ++i) a += C3[j * 2 + i] * g[i]; c[j] = a; }
  }
  float wn[4];
  for (int q = 0; q < 4; ++q) {
    red[k] = srcv ? l[q] : -3.0e38f;
    __syncthreads();
    for (int off = 128; off > 0; off >>= 1) { if (k < off) red[k] = fmaxf(red[k], red[k + off]); __syncthreads(); }
    float mx = red[0];
    __syncthreads();
    float ev = (srcv && melv) ? __expf(l[q] - mx) : 0.f;
    red[k] = ev;
    __syncthreads();
    for (int off = 128; off > 0; off >>= 1) { if (k < off) red[k] += red[k + off]; __syncthreads(); }
    float sm = red[0];
    __syncthreads();
    wn[q] = melv ? (ev / sm) : 0.f;
  }
#pragma unroll
  for (int q = 0; q < 4; ++q)
#pragma unroll
    for (int p = 0; p < 2; ++p) atomicAdd(&BcS[q * 2 + p], wn[q] * c[p]);

  size_t wb = (((size_t)b * Tm + t) * Ts + k) * 4;
#pragma unroll
  for (int q = 0; q < 4; ++q) outW[wb + q] = wn[q];
  size_t wf = (size_t)b * TmP * KQ + (size_t)t * KQ + k * 4;
#pragma unroll
  for (int q = 0; q < 4; ++q) Wbf[wf + q] = f2bf(wn[q]);
  __syncthreads();
  if (k < 8) Bc[((size_t)b * Tm + t) * 8 + k] = BcS[k];
}

// ---------------------------------------------------------------------------
// K5: G^T[b][o, k*4+q] = sum_h V[b,k,h] * lw_w[o, q*192+h]  (WMMA)
// ---------------------------------------------------------------------------
__global__ void gemm_g_kernel(const unsigned short* __restrict__ Vbf,
                              const unsigned short* __restrict__ lwbf,
                              unsigned short* __restrict__ Gt) {
  int tid = threadIdx.x;
  int wave = tid >> 5, lane = tid & 31;
  int mt = blockIdx.x;
  int nt = blockIdx.y * 4 + wave;
  int bz = blockIdx.z;
  int b = bz >> 2, q = bz & 3;
  const unsigned short* A  = Vbf + (size_t)b * Ts * Dd + (size_t)(mt * 16) * Dd;
  const unsigned short* Bm = lwbf + (size_t)(nt * 16) * (4 * Dd) + q * Dd;
  v8f acc = {};
  for (int kk = 0; kk < Dd; kk += 32) {
    v16bf af = load_fragA(A, Dd, lane, kk);
    v16bf bf = load_fragB(Bm, 4 * Dd, lane, kk);
    acc = wmma_bf16(af, bf, acc);
  }
  int o = nt * 16 + (lane & 15);
  int kr0 = mt * 16 + (lane >> 4) * 8;
#pragma unroll
  for (int r = 0; r < 8; ++r)
    Gt[(size_t)b * Dd * KQ + (size_t)o * KQ + (kr0 + r) * 4 + q] = f2bf(acc[r]);
}

// ---------------------------------------------------------------------------
// K3: up = Wflat @ G^T  (M=1008, N=192, K=1024, WMMA).
//     B-panel (192 x KC bf16) is DMA'd into LDS by the Tensor Data Mover
//     (tensor_load_to_lds, TENSORcnt), A streams from HBM as b128 clauses.
//     Fused epilogue: + lw_b + le_b + Bc@le^T, LayerNorm, mel-mask, bf16 store.
// ---------------------------------------------------------------------------
__global__ void gemm_up_kernel(const unsigned short* __restrict__ Wbf,
                               const unsigned short* __restrict__ Gt,
                               const float* __restrict__ Bc,
                               const float* __restrict__ le_w,
                               const float* __restrict__ lw_b,
                               const float* __restrict__ le_b,
                               const float* __restrict__ ln_g,
                               const float* __restrict__ ln_b,
                               const int* __restrict__ mel_len_i,
                               unsigned short* __restrict__ upbf) {
  int tid = threadIdx.x;
  int wave = tid >> 5, lane = tid & 31;
  int m0 = blockIdx.x * 16;
  int b = blockIdx.y;
  __shared__ unsigned short bstage[Dd * KC];   // 192 rows x 64-K panel (24 KB)
  __shared__ float ups[16 * Dd];
  __shared__ float sLe[Dd * 8];
  __shared__ float sBias[Dd];
  __shared__ float sLnG[Dd], sLnB[Dd];
  __shared__ float sBc[16 * 8];
  __shared__ float sMu[16], sRs[16];

  for (int i = tid; i < Dd * 8; i += 128) sLe[i] = le_w[i];
  for (int i = tid; i < Dd; i += 128) {
    sBias[i] = lw_b[i] + le_b[i];
    sLnG[i] = ln_g[i];
    sLnB[i] = ln_b[i];
  }
  { int row = tid >> 3, f = tid & 7; int t = m0 + row;
    sBc[tid] = (t < Tm) ? Bc[((size_t)b * Tm + t) * 8 + f] : 0.f; }

  const unsigned short* A  = Wbf + (size_t)b * TmP * KQ + (size_t)m0 * KQ;
  const unsigned short* Bm = Gt + (size_t)b * Dd * KQ;
  unsigned ldsOff = (unsigned)(size_t)(&bstage[0]);
  v8f acc[3] = {};

  for (int kk0 = 0; kk0 < KQ; kk0 += KC) {
    __syncthreads();                           // previous panel fully consumed
    if (tid < 32) {
      // Tensor DMA descriptor (D# group0 + group1), 2-D tensor:
      //   tensor: [dim1=192 rows, dim0=1024 K] bf16, row stride 1024
      //   tile:   [192, KC] at column kk0  ->  LDS, rows packed (ld = KC)
      unsigned long long ga =
          (unsigned long long)(size_t)Bm + (unsigned long long)kk0 * 2ull;
      v4u g0;
      g0[0] = 1u;                                        // count=1, user D#
      g0[1] = ldsOff;                                    // lds_addr
      g0[2] = (unsigned)(ga & 0xffffffffull);            // global_addr[31:0]
      g0[3] = (unsigned)((ga >> 32) & 0x01ffffffull)     // global_addr[56:32]
              | (2u << 30);                              // type = 2 (image)
      v8u g1;
      g1[0] = (1u << 16);                 // data_size = 1 (2 bytes)
      g1[1] = ((unsigned)KQ) << 16;       // tensor_dim0 = 1024
      g1[2] = ((unsigned)Dd) << 16;       // tensor_dim1 = 192
      g1[3] = ((unsigned)KC) << 16;       // tile_dim0   = KC
      g1[4] = (unsigned)Dd;               // tile_dim1   = 192
      g1[5] = (unsigned)KQ;               // tensor_dim0_stride = 1024
      g1[6] = 0u;
      g1[7] = 0u;
      asm volatile("tensor_load_to_lds %0, %1" : : "s"(g0), "s"(g1) : "memory");
      __builtin_amdgcn_s_wait_tensorcnt(0);
    }
    __syncthreads();                           // panel visible to all waves

#pragma unroll
    for (int kk = 0; kk < KC; kk += 32) {
      __builtin_prefetch(A + (size_t)(lane & 15) * KQ + kk0 + kk + 128, 0, 0);
      v16bf af = load_fragA(A, KQ, lane, kk0 + kk);
#pragma unroll
      for (int j = 0; j < 3; ++j) {
        int nt = wave * 3 + j;
        v16bf bf = load_fragB(&bstage[(size_t)(nt * 16) * KC], KC, lane, kk);
        acc[j] = wmma_bf16(af, bf, acc[j]);
      }
    }
  }
  int r0 = (lane >> 4) * 8, c0 = lane & 15;
#pragma unroll
  for (int j = 0; j < 3; ++j) {
    int nt = wave * 3 + j;
#pragma unroll
    for (int r = 0; r < 8; ++r)
      ups[(r0 + r) * Dd + nt * 16 + c0] = acc[j][r];
  }
  __syncthreads();
  for (int idx = tid; idx < 16 * Dd; idx += 128) {
    int row = idx / Dd, o = idx - row * Dd;
    float v = ups[idx] + sBias[o];
    float a2 = 0.f;
#pragma unroll
    for (int f = 0; f < 8; ++f) a2 += sBc[row * 8 + f] * sLe[o * 8 + f];
    ups[idx] = v + a2;
  }
  __syncthreads();
  if (tid < 16) {
    float su = 0.f, sq = 0.f;
    for (int o = 0; o < Dd; ++o) { float v = ups[tid * Dd + o]; su += v; sq += v * v; }
    float mu = su * (1.f / Dd);
    sMu[tid] = mu;
    sRs[tid] = rsqrtf(sq * (1.f / Dd) - mu * mu + EPSL);
  }
  __syncthreads();
  int ml = mel_len_i[b];
  for (int idx = tid; idx < 16 * Dd; idx += 128) {
    int row = idx / Dd, o = idx - row * Dd;
    int t = m0 + row;
    float v = (t < ml) ? (ups[idx] - sMu[row]) * sRs[row] * sLnG[o] + sLnB[o] : 0.f;
    upbf[(size_t)b * TmP * Dd + (size_t)t * Dd + o] = f2bf(v);
  }
}

// ---------------------------------------------------------------------------
// K4: out = up @ po^T + po_b  (M=1008, N=192, K=192, WMMA), store t<1000
// ---------------------------------------------------------------------------
__global__ void gemm_out_kernel(const unsigned short* __restrict__ upbf,
                                const unsigned short* __restrict__ pobf,
                                const float* __restrict__ po_b,
                                float* __restrict__ outp) {
  int tid = threadIdx.x;
  int wave = tid >> 5, lane = tid & 31;
  int m0 = blockIdx.x * 16;
  int b = blockIdx.y;
  const unsigned short* A = upbf + (size_t)b * TmP * Dd + (size_t)m0 * Dd;
  v8f acc[3] = {};
  for (int kk = 0; kk < Dd; kk += 32) {
    v16bf af = load_fragA(A, Dd, lane, kk);
#pragma unroll
    for (int j = 0; j < 3; ++j) {
      int nt = wave * 3 + j;
      v16bf bf = load_fragB(pobf + (size_t)(nt * 16) * Dd, Dd, lane, kk);
      acc[j] = wmma_bf16(af, bf, acc[j]);
    }
  }
  int r0 = (lane >> 4) * 8, c0 = lane & 15;
#pragma unroll
  for (int j = 0; j < 3; ++j) {
    int o = (wave * 3 + j) * 16 + c0;
    float pb = po_b[o];
#pragma unroll
    for (int r = 0; r < 8; ++r) {
      int t = m0 + r0 + r;
      if (t < Tm) outp[((size_t)b * Tm + t) * Dd + o] = acc[j][r] + pb;
    }
  }
}

// ---------------------------------------------------------------------------
extern "C" void kernel_launch(void* const* d_in, const int* in_sizes, int n_in,
                              void* d_out, int out_size, void* d_ws, size_t ws_size,
                              hipStream_t stream) {
  (void)in_sizes; (void)n_in; (void)out_size; (void)ws_size;
  const float* dur     = (const float*)d_in[0];
  const float* V       = (const float*)d_in[1];
  const int*   src_len = (const int*)d_in[2];
  // d_in[3]=src_mask (derived from src_len), d_in[4]=max_src_len : unused
  const float* cwW  = (const float*)d_in[5];
  const float* cwB  = (const float*)d_in[6];
  const float* cwM  = (const float*)d_in[7];
  const float* cwVv = (const float*)d_in[8];
  const float* cwG  = (const float*)d_in[9];
  const float* cwBB = (const float*)d_in[10];
  const float* cwLG = (const float*)d_in[11];
  const float* cwLB = (const float*)d_in[12];
  const float* swW1 = (const float*)d_in[13];
  const float* swB1 = (const float*)d_in[14];
  const float* swW2 = (const float*)d_in[15];
  const float* swB2 = (const float*)d_in[16];
  const float* swW3 = (const float*)d_in[17];
  const float* swB3 = (const float*)d_in[18];
  const float* ccW  = (const float*)d_in[19];
  const float* ccB  = (const float*)d_in[20];
  const float* ccM  = (const float*)d_in[21];
  const float* ccVv = (const float*)d_in[22];
  const float* ccG  = (const float*)d_in[23];
  const float* ccBB = (const float*)d_in[24];
  const float* ccLG = (const float*)d_in[25];
  const float* ccLB = (const float*)d_in[26];
  const float* scW1 = (const float*)d_in[27];
  const float* scB1 = (const float*)d_in[28];
  const float* scW2 = (const float*)d_in[29];
  const float* scB2 = (const float*)d_in[30];
  const float* scW3 = (const float*)d_in[31];
  const float* scB3 = (const float*)d_in[32];
  const float* lw_w = (const float*)d_in[33];
  const float* lw_b = (const float*)d_in[34];
  const float* le_w = (const float*)d_in[35];
  const float* le_b = (const float*)d_in[36];
  const float* ln_g = (const float*)d_in[37];
  const float* ln_b = (const float*)d_in[38];
  const float* po_w = (const float*)d_in[39];
  const float* po_b = (const float*)d_in[40];

  float* outp    = (float*)d_out;
  float* outMask = outp + (size_t)Bn * Tm * Dd;
  float* outMlen = outMask + (size_t)Bn * Tm;
  float* outW    = outMlen + Bn;

  size_t off = 0;
  auto alloc = [&](size_t bytes) {
    off = (off + 255) & ~(size_t)255;
    size_t o = off; off += bytes; return o;
  };
  char* w = (char*)d_ws;
  float* e_k  = (float*)(w + alloc((size_t)Bn * Ts * 4));
  float* s_k  = (float*)(w + alloc((size_t)Bn * Ts * 4));
  int*   mli  = (int*)  (w + alloc((size_t)Bn * 4));
  float* VcW  = (float*)(w + alloc((size_t)Bn * Ts * CO * 4));
  float* VcC  = (float*)(w + alloc((size_t)Bn * Ts * CO * 4));
  unsigned short* Vbf  = (unsigned short*)(w + alloc((size_t)Bn * Ts * Dd * 2));
  unsigned short* lwbf = (unsigned short*)(w + alloc((size_t)Dd * 4 * Dd * 2));
  unsigned short* pobf = (unsigned short*)(w + alloc((size_t)Dd * Dd * 2));
  unsigned short* Wbf  = (unsigned short*)(w + alloc((size_t)Bn * TmP * KQ * 2));
  unsigned short* Gt   = (unsigned short*)(w + alloc((size_t)Bn * Dd * KQ * 2));
  float* Bc            = (float*)         (w + alloc((size_t)Bn * Tm * 8 * 4));
  unsigned short* upbf = (unsigned short*)(w + alloc((size_t)Bn * TmP * Dd * 2));

  prep_kernel<<<dim3(Bn), dim3(Ts), 0, stream>>>(
      dur, V, cwW, cwB, cwM, cwVv, cwG, cwBB, cwLG, cwLB,
      ccW, ccB, ccM, ccVv, ccG, ccBB, ccLG, ccLB,
      e_k, s_k, mli, VcW, VcC, Vbf, outMask, outMlen);

  cvt_w_kernel<<<dim3(72), dim3(256), 0, stream>>>(lw_w, po_w, lwbf, pobf);

  mlp_kernel<<<dim3(Tm, Bn), dim3(Ts), 0, stream>>>(
      e_k, s_k, mli, src_len, VcW, VcC,
      swW1, swB1, swW2, swB2, swW3, swB3,
      scW1, scB1, scW2, scB2, scW3, scB3,
      outW, Wbf, Bc);

  gemm_g_kernel<<<dim3(Ts / 16, 3, Bn * 4), dim3(128), 0, stream>>>(Vbf, lwbf, Gt);

  gemm_up_kernel<<<dim3(TmP / 16, Bn), dim3(128), 0, stream>>>(
      Wbf, Gt, Bc, le_w, lw_b, le_b, ln_g, ln_b, mli, upbf);

  gemm_out_kernel<<<dim3(TmP / 16, Bn), dim3(128), 0, stream>>>(
      upbf, pobf, po_b, outp);
}
